// KMCluster_24962349924819
// MI455X (gfx1250) — matrix operand order repaired and checked by reading, not compiled, not run on hardware
//
#include <hip/hip_runtime.h>
#include <hip/hip_bf16.h>

#define N_PTS 131072
#define DIMS  256
#define KCL   128
#define N_ITERS 10
#define ACC_BLOCKS 256
#define PTS_PER_BLK (N_PTS / ACC_BLOCKS)   // 512

typedef __attribute__((ext_vector_type(16))) __bf16 v16bf;
typedef __attribute__((ext_vector_type(8)))  __bf16 v8bf;
typedef __attribute__((ext_vector_type(8)))  float  v8f;
typedef __attribute__((ext_vector_type(4)))  float  v4f;

union V16U { v16bf v; v8bf h[2]; };

// ---------------------------------------------------------------------------
// Assignment: one wave -> 16 points x 128 clusters via bf16-split WMMA.
// dist(p,c) ~ ||c||^2 - 2 * x_p . c   (x^2 term dropped: constant per point)
// x.c computed as xhi.chi + xhi.clo + xlo.chi in f32 accumulators.
// ---------------------------------------------------------------------------
template <bool FINAL>
__global__ __launch_bounds__(256) void assign_kernel(
    const float*  __restrict__ X,
    const __bf16* __restrict__ Chi,
    const __bf16* __restrict__ Clo,
    const float*  __restrict__ c2,
    int*          __restrict__ out_i,
    float*        __restrict__ out_f)
{
    const int lane  = threadIdx.x & 31;
    const int wave  = threadIdx.x >> 5;
    const int lr    = lane & 15;
    const bool hi   = lane >= 16;
    const int pbase = (blockIdx.x * 8 + wave) * 16;

    v8f zero = {0.f, 0.f, 0.f, 0.f, 0.f, 0.f, 0.f, 0.f};
    v8f acc[8];
#pragma unroll
    for (int m = 0; m < 8; ++m) acc[m] = zero;

    const float* xrow = X + (size_t)(pbase + lr) * DIMS;

#pragma unroll
    for (int k = 0; k < 8; ++k) {
        // ---- B fragment: 16 consecutive K-values of this lane's point ----
        // B layout (32x16 bf16): lanes 0-15 -> K = 32k+0..15, lanes 16-31 -> K = 32k+16..31
        const float* xp = xrow + 32 * k + (hi ? 16 : 0);
        float xv[16];
#pragma unroll
        for (int q = 0; q < 4; ++q) {
            v4f t = *reinterpret_cast<const v4f*>(xp + 4 * q);
#pragma unroll
            for (int j = 0; j < 4; ++j) xv[4 * q + j] = t[j];
        }
        v16bf bh, bl;
#pragma unroll
        for (int e = 0; e < 16; ++e) {
            float x  = xv[e];
            __bf16 h = (__bf16)x;           // RTNE
            bh[e]    = h;
            bl[e]    = (__bf16)(x - (float)h);
        }

        // A layout (16x32 bf16): row M = lane%16 of cluster tile,
        // kb = 16 for lanes>=16; e0-15 = K kb..kb+15
        const int kbA = 32 * k + (hi ? 16 : 0);
#pragma unroll
        for (int m = 0; m < 8; ++m) {
            const __bf16* ah = Chi + (size_t)(m * 16 + lr) * DIMS + kbA;
            const __bf16* al = Clo + (size_t)(m * 16 + lr) * DIMS + kbA;
            V16U Ahi, Alo;
            Ahi.h[0] = *reinterpret_cast<const v8bf*>(ah);
            Ahi.h[1] = *reinterpret_cast<const v8bf*>(ah + 8);
            Alo.h[0] = *reinterpret_cast<const v8bf*>(al);
            Alo.h[1] = *reinterpret_cast<const v8bf*>(al + 8);

            acc[m] = __builtin_amdgcn_wmma_f32_16x16x32_bf16(
                false, Ahi.v, false, bh, (short)0, acc[m], false, false);
            acc[m] = __builtin_amdgcn_wmma_f32_16x16x32_bf16(
                false, Ahi.v, false, bl, (short)0, acc[m], false, false);
            acc[m] = __builtin_amdgcn_wmma_f32_16x16x32_bf16(
                false, Alo.v, false, bh, (short)0, acc[m], false, false);
        }
    }

    // D layout: VGPR r -> cluster row (m*16 + r + 8*hi), column N = lane%16 (point)
    float best = 3.4e38f;
    int   bidx = 0;
#pragma unroll
    for (int m = 0; m < 8; ++m) {
#pragma unroll
        for (int r = 0; r < 8; ++r) {
            int   cid = m * 16 + (hi ? 8 : 0) + r;
            float d   = c2[cid] - 2.0f * acc[m][r];
            if (d < best || (d == best && cid < bidx)) { best = d; bidx = cid; }
        }
    }
    // merge the two half-wave lanes covering the same point (disjoint clusters)
    float ob = __shfl_xor(best, 16, 32);
    int   oi = __shfl_xor(bidx, 16, 32);
    if (ob < best || (ob == best && oi < bidx)) { best = ob; bidx = oi; }

    if (lane < 16) {
        int p = pbase + lr;
        if (FINAL) out_f[p] = (float)bidx;
        else       out_i[p] = bidx;
    }
}

// ---------------------------------------------------------------------------
// Segment-sum with LDS privatization: thread `tid` exclusively owns dim
// column `tid` of s_sums[cluster][dim] -> plain LDS RMW, no atomics, no bank
// conflicts. One coalesced global atomicAdd per cell at flush.
// ---------------------------------------------------------------------------
__global__ __launch_bounds__(256) void accum_kernel(
    const float* __restrict__ X,
    const int*   __restrict__ assign,
    float*       __restrict__ sums,
    float*       __restrict__ counts)
{
    __shared__ float s_sums[KCL * DIMS];   // 128 KB (<= 320 KB/WGP)
    __shared__ float s_cnt[KCL];
    const int tid = threadIdx.x;

    for (int i = tid; i < KCL * DIMS; i += 256) s_sums[i] = 0.f;
    if (tid < KCL) s_cnt[tid] = 0.f;
    __syncthreads();

    const int p0 = blockIdx.x * PTS_PER_BLK;
    for (int p = p0; p < p0 + PTS_PER_BLK; p += 4) {
        int a0 = assign[p + 0], a1 = assign[p + 1];
        int a2 = assign[p + 2], a3 = assign[p + 3];
        float v0 = X[(size_t)(p + 0) * DIMS + tid];
        float v1 = X[(size_t)(p + 1) * DIMS + tid];
        float v2 = X[(size_t)(p + 2) * DIMS + tid];
        float v3 = X[(size_t)(p + 3) * DIMS + tid];
        s_sums[a0 * DIMS + tid] += v0;
        s_sums[a1 * DIMS + tid] += v1;
        s_sums[a2 * DIMS + tid] += v2;
        s_sums[a3 * DIMS + tid] += v3;
        if (tid == 0) {
            s_cnt[a0] += 1.f; s_cnt[a1] += 1.f;
            s_cnt[a2] += 1.f; s_cnt[a3] += 1.f;
        }
    }
    __syncthreads();

#pragma unroll 4
    for (int c = 0; c < KCL; ++c)
        atomicAdd(&sums[c * DIMS + tid], s_sums[c * DIMS + tid]);
    if (tid < KCL) atomicAdd(&counts[tid], s_cnt[tid]);
}

// ---------------------------------------------------------------------------
// Centroid update (or init from X[:128]); also emits bf16 hi/lo split and
// per-cluster ||c||^2 (block reduction). One block per cluster.
// ---------------------------------------------------------------------------
template <bool INIT>
__global__ __launch_bounds__(256) void update_kernel(
    const float* __restrict__ X,
    const float* __restrict__ sums,
    const float* __restrict__ counts,
    float*       __restrict__ C,
    __bf16*      __restrict__ Chi,
    __bf16*      __restrict__ Clo,
    float*       __restrict__ c2)
{
    const int c = blockIdx.x;
    const int d = threadIdx.x;
    float nc;
    if (INIT) {
        nc = X[(size_t)c * DIMS + d];
    } else {
        float cnt = counts[c];
        nc = (cnt > 0.f) ? sums[c * DIMS + d] / cnt : C[c * DIMS + d];
    }
    C[c * DIMS + d] = nc;
    __bf16 h = (__bf16)nc;
    Chi[c * DIMS + d] = h;
    Clo[c * DIMS + d] = (__bf16)(nc - (float)h);

    __shared__ float red[256];
    red[d] = nc * nc;
    __syncthreads();
    for (int s = 128; s > 0; s >>= 1) {
        if (d < s) red[d] += red[d + s];
        __syncthreads();
    }
    if (d == 0) c2[c] = red[0];
}

__global__ void zero_kernel(float* __restrict__ sums, float* __restrict__ counts)
{
    int i = blockIdx.x * blockDim.x + threadIdx.x;
    if (i < KCL * DIMS) sums[i] = 0.f;
    if (i < KCL)        counts[i] = 0.f;
}

// ---------------------------------------------------------------------------
extern "C" void kernel_launch(void* const* d_in, const int* in_sizes, int n_in,
                              void* d_out, int out_size, void* d_ws, size_t ws_size,
                              hipStream_t stream)
{
    (void)in_sizes; (void)n_in; (void)out_size; (void)ws_size;
    const float* X = (const float*)d_in[0];

    // workspace carve (all offsets multiples of 512B -> vector-load aligned)
    char* w = (char*)d_ws;
    float* C      = (float*)w;  w += KCL * DIMS * sizeof(float);   // 128 KB
    float* sums   = (float*)w;  w += KCL * DIMS * sizeof(float);   // 128 KB
    float* counts = (float*)w;  w += 512;
    float* c2     = (float*)w;  w += 512;
    __bf16* Chi   = (__bf16*)w; w += KCL * DIMS * sizeof(__bf16);  // 64 KB
    __bf16* Clo   = (__bf16*)w; w += KCL * DIMS * sizeof(__bf16);  // 64 KB
    int* assign   = (int*)w;    w += (size_t)N_PTS * sizeof(int);  // 512 KB

    // init centroids = X[:128]
    update_kernel<true><<<KCL, 256, 0, stream>>>(X, sums, counts, C, Chi, Clo, c2);

    for (int it = 0; it < N_ITERS; ++it) {
        assign_kernel<false><<<N_PTS / 128, 256, 0, stream>>>(
            X, Chi, Clo, c2, assign, nullptr);
        zero_kernel<<<(KCL * DIMS + 255) / 256, 256, 0, stream>>>(sums, counts);
        accum_kernel<<<ACC_BLOCKS, 256, 0, stream>>>(X, assign, sums, counts);
        update_kernel<false><<<KCL, 256, 0, stream>>>(X, sums, counts, C, Chi, Clo, c2);
    }

    // final predictions with the 10-iteration centroids
    assign_kernel<true><<<N_PTS / 128, 256, 0, stream>>>(
        X, Chi, Clo, c2, nullptr, (float*)d_out);
}
